// multiheadlatentattention_18528488915353
// MI455X (gfx1250) — matrix operand order repaired and checked by reading, not compiled, not run on hardware
//
#include <hip/hip_runtime.h>
#include <cstdint>

// ---------------------------------------------------------------------------
// MLA forward for MI455X (gfx1250, wave32, WMMA).
// All projections + attention run as bf16 WMMA with fp32 accumulation.
// Intermediates (ckv, q_lat, key, query, value, attn_out) are bf16 in d_ws.
// ---------------------------------------------------------------------------

#define D_IN     2048
#define D_OUT    2048
#define DL       512
#define NUM_HEAD 16
#define HEAD_DIM 128
#define CTX      2048
#define BATCH    2
#define NROW     (BATCH * CTX)   // 4096

// gfx1250 async global->LDS path (guarded: falls back to plain b128 copies)
#if defined(__AMDGCN__) && __has_builtin(__builtin_amdgcn_global_load_async_to_lds_b128) && __has_builtin(__builtin_amdgcn_s_wait_asynccnt)
#define USE_ASYNC_LDS 1
#else
#define USE_ASYNC_LDS 0
#endif

typedef __bf16 v16bf __attribute__((ext_vector_type(16)));
typedef float  v8f   __attribute__((ext_vector_type(8)));

union Frag16 {            // 16 bf16 = 32 bytes = two 16B chunks
  v16bf v;
  uint4 q[2];
};

#if USE_ASYNC_LDS
// Builtin signature (from hipcc diagnostic): first arg is
//   'int __vector(4) __device__ *'  (v4i in AS1), second is the LDS side.
// Produce AS-qualified pointers via uintptr_t (CK-style): global flat value
// equals AS1 value; LDS flat address truncates to the AS3 offset (ISA 10.2).
typedef int gv4i __attribute__((vector_size(16)));
__device__ __forceinline__ void async_b128_to_lds(unsigned short* lds_dst,
                                                  const unsigned short* gsrc) {
  __builtin_amdgcn_global_load_async_to_lds_b128(
      reinterpret_cast<gv4i __attribute__((address_space(1)))*>(
          reinterpret_cast<uintptr_t>(gsrc)),
      reinterpret_cast<gv4i __attribute__((address_space(3)))*>(
          reinterpret_cast<uintptr_t>(lds_dst)),
      0, 0);
}
#endif

__device__ __forceinline__ unsigned short f2bf(float f) {
  unsigned int u = __builtin_bit_cast(unsigned int, f);
  u += 0x7FFFu + ((u >> 16) & 1u);           // round-to-nearest-even
  return (unsigned short)(u >> 16);
}

__device__ __forceinline__ void store_c(float* p, float v)          { *p = v; }
__device__ __forceinline__ void store_c(unsigned short* p, float v) { *p = f2bf(v); }

// stage 4 consecutive A elements (convert if fp32, raw copy if bf16)
__device__ __forceinline__ void stage_a4(unsigned short* dst, const float* src) {
  float4 f = *(const float4*)src;
  dst[0] = f2bf(f.x); dst[1] = f2bf(f.y); dst[2] = f2bf(f.z); dst[3] = f2bf(f.w);
}
__device__ __forceinline__ void stage_a4(unsigned short* dst, const unsigned short* src) {
  *(uint2*)dst = *(const uint2*)src;
}

// ---------------------------------------------------------------------------
// Tiled GEMM:  C[M x N] (at col_off) = A[M x K] @ W[K x N] + bias[N]
//   A: fp32 or bf16 (row-major, lda) ; W: fp32 (row-major, ldw)
//   C: bf16 or fp32 (ldc, col offset col_off)
// Block: 256 threads (8 waves). WG tile 64x256, wave tile 32x64, K step 32.
// 8 WMMAs per wave per K-step; LDS layouts make fragment loads contiguous
// 16B/32B reads matching the CDNA5 16-bit A / B VGPR layouts.
// ---------------------------------------------------------------------------
#define AS_STRIDE 48   // 64 rows x 32 K   (pad keeps 16B alignment)
#define BS_STRIDE 48   // 256 n-rows x 32 K (transposed W)

template <typename AT, typename CT>
__global__ __launch_bounds__(256)
void gemm_wmma_bf16(const AT* __restrict__ A, int lda,
                    const float* __restrict__ W, int ldw,
                    const float* __restrict__ bias,
                    CT* __restrict__ C, int ldc, int col_off,
                    int M, int N, int K)
{
  __shared__ __align__(16) unsigned short As[64 * AS_STRIDE];
  __shared__ __align__(16) unsigned short Bs[256 * BS_STRIDE];

  const int t    = threadIdx.x;
  const int lane = t & 31;
  const int wv   = t >> 5;        // 0..7
  const int wr   = wv >> 2;       // 0..1  (row group of 32)
  const int wc   = wv & 3;        // 0..3  (col group of 64)
  const int bm   = blockIdx.x * 64;
  const int bn   = blockIdx.y * 256;
  const int l15  = lane & 15;
  const int half = lane >> 4;

  v8f acc[2][4];
#pragma unroll
  for (int i = 0; i < 2; ++i)
#pragma unroll
    for (int j = 0; j < 4; ++j) acc[i][j] = v8f{};

  for (int k0 = 0; k0 < K; k0 += 32) {
    // ---- stage A tile (64 x 32) as bf16, row-major, 4 elems per move ----
    for (int e = t; e < 64 * 8; e += 256) {
      int r = e >> 3, c4 = e & 7;
      stage_a4(&As[r * AS_STRIDE + c4 * 4], &A[(size_t)(bm + r) * lda + k0 + c4 * 4]);
    }
    // ---- stage W tile (32 x 256) transposed: Bs[n][k], float4 reads ----
    for (int e = t; e < 32 * 64; e += 256) {
      int kk = e >> 6, n4 = e & 63;
      float4 f = *(const float4*)&W[(size_t)(k0 + kk) * ldw + bn + n4 * 4];
      Bs[(n4 * 4 + 0) * BS_STRIDE + kk] = f2bf(f.x);
      Bs[(n4 * 4 + 1) * BS_STRIDE + kk] = f2bf(f.y);
      Bs[(n4 * 4 + 2) * BS_STRIDE + kk] = f2bf(f.z);
      Bs[(n4 * 4 + 3) * BS_STRIDE + kk] = f2bf(f.w);
    }
    // ---- prefetch next K tile (global_prefetch_b8) ----
    if (k0 + 32 < K) {
      __builtin_prefetch(&A[(size_t)(bm + (t >> 3)) * lda + k0 + 32 + (t & 7) * 4], 0, 1);
      __builtin_prefetch(&W[(size_t)(k0 + 32 + (t >> 6)) * ldw + bn + (t & 63) * 4], 0, 1);
    }
    __syncthreads();

    // ---- load fragments ----
    Frag16 af[2], bf[4];
#pragma unroll
    for (int i = 0; i < 2; ++i) {
      int row = wr * 32 + i * 16 + l15;
      const unsigned short* p = &As[row * AS_STRIDE];
      af[i].q[0] = *(const uint4*)(p + half * 8);        // K {0..7} / {8..15}
      af[i].q[1] = *(const uint4*)(p + 16 + half * 8);   // K {16..23} / {24..31}
    }
#pragma unroll
    for (int j = 0; j < 4; ++j) {
      int col = wc * 64 + j * 16 + l15;
      const unsigned short* p = &Bs[col * BS_STRIDE];
      bf[j].q[0] = *(const uint4*)(p + half * 8);        // K {0..7} / {8..15}
      bf[j].q[1] = *(const uint4*)(p + 16 + half * 8);   // K {16..23} / {24..31}
    }

#pragma unroll
    for (int i = 0; i < 2; ++i)
#pragma unroll
      for (int j = 0; j < 4; ++j)
        acc[i][j] = __builtin_amdgcn_wmma_f32_16x16x32_bf16(
            false, af[i].v, false, bf[j].v, (short)0, acc[i][j], false, false);

    __syncthreads();
  }

  // ---- epilogue: C-fragment layout -> global ----
#pragma unroll
  for (int i = 0; i < 2; ++i) {
#pragma unroll
    for (int j = 0; j < 4; ++j) {
      int ncol = wc * 64 + j * 16 + l15;
      float bv = bias[bn + ncol];
#pragma unroll
      for (int e = 0; e < 8; ++e) {
        int m = bm + wr * 32 + i * 16 + half * 8 + e;
        float v = acc[i][j][e] + bv;
        store_c(&C[(size_t)m * ldc + col_off + bn + ncol], v);
      }
    }
  }
}

// ---------------------------------------------------------------------------
// Fused causal attention (flash-style, online softmax).
// Grid: (CTX/64, NUM_HEAD, BATCH). Block: 128 threads = 4 waves.
// Each wave owns 16 q-rows; Q fragments stay resident in VGPRs.
// K tile is staged with global_load_async_to_lds_b128 when available.
// ---------------------------------------------------------------------------
#define KS_STRIDE 144   // 64 key-rows x 128 d
#define VT_STRIDE 80    // 128 d-rows x 64 keys (V transposed)
#define PS_STRIDE 72    // 16 q-rows x 64 keys per wave

__global__ __launch_bounds__(128)
void mla_attention(const unsigned short* __restrict__ query,
                   const unsigned short* __restrict__ key,
                   const unsigned short* __restrict__ value,
                   unsigned short* __restrict__ attnout)
{
  __shared__ __align__(16) unsigned short Ks[64 * KS_STRIDE];
  __shared__ __align__(16) unsigned short Vt[128 * VT_STRIDE];
  __shared__ __align__(16) unsigned short Ps[4][16 * PS_STRIDE];

  const int qblk = blockIdx.x;        // 0..31
  const int h    = blockIdx.y;        // 0..15
  const int b    = blockIdx.z;        // 0..1
  const int t    = threadIdx.x;
  const int lane = t & 31;
  const int wv   = t >> 5;            // 0..3
  const int l15  = lane & 15;
  const int half = lane >> 4;

  const int rowbase = b * CTX;             // batch offset in 4096-row space
  const int qrow0   = qblk * 64 + wv * 16; // this wave's first q row (in-batch)
  const float scale = 0.08838834764831845f; // 1/sqrt(128)

  // ---- Q fragments: 16 rows x 128 d, 4 K-chunks of 32, resident in regs ----
  Frag16 qf[4];
  {
    const unsigned short* qp =
        query + (size_t)(rowbase + qrow0 + l15) * D_OUT + h * HEAD_DIM;
#pragma unroll
    for (int c = 0; c < 4; ++c) {
      qf[c].q[0] = *(const uint4*)(qp + c * 32 + half * 8);
      qf[c].q[1] = *(const uint4*)(qp + c * 32 + 16 + half * 8);
    }
  }

  v8f o[8];
#pragma unroll
  for (int i = 0; i < 8; ++i) o[i] = v8f{};
  float rmax[8], rsum[8];
#pragma unroll
  for (int e = 0; e < 8; ++e) { rmax[e] = -INFINITY; rsum[e] = 0.0f; }

  const int nkb = qblk + 1;   // causal: key blocks 0..qblk
  for (int kb = 0; kb < nkb; ++kb) {
    __syncthreads();   // previous iteration's LDS reads complete

    // ---- stage K block (64 x 128) row-major bf16, 16B chunks ----
#if USE_ASYNC_LDS
    for (int e = t; e < 64 * 16; e += 128) {
      int r = e >> 4, g = e & 15;
      async_b128_to_lds(&Ks[r * KS_STRIDE + g * 8],
                        key + (size_t)(rowbase + kb * 64 + r) * D_OUT +
                            h * HEAD_DIM + g * 8);
    }
#else
    for (int e = t; e < 64 * 16; e += 128) {
      int r = e >> 4, g = e & 15;
      *(uint4*)&Ks[r * KS_STRIDE + g * 8] =
          *(const uint4*)(key + (size_t)(rowbase + kb * 64 + r) * D_OUT +
                          h * HEAD_DIM + g * 8);
    }
#endif
    // ---- stage V block transposed: Vt[d][keyrow], 8 bf16 per read ----
    for (int e = t; e < 64 * 16; e += 128) {
      int r = e >> 4, g = e & 15;
      uint4 d = *(const uint4*)(value + (size_t)(rowbase + kb * 64 + r) * D_OUT +
                                h * HEAD_DIM + g * 8);
      Vt[(g * 8 + 0) * VT_STRIDE + r] = (unsigned short)(d.x & 0xffffu);
      Vt[(g * 8 + 1) * VT_STRIDE + r] = (unsigned short)(d.x >> 16);
      Vt[(g * 8 + 2) * VT_STRIDE + r] = (unsigned short)(d.y & 0xffffu);
      Vt[(g * 8 + 3) * VT_STRIDE + r] = (unsigned short)(d.y >> 16);
      Vt[(g * 8 + 4) * VT_STRIDE + r] = (unsigned short)(d.z & 0xffffu);
      Vt[(g * 8 + 5) * VT_STRIDE + r] = (unsigned short)(d.z >> 16);
      Vt[(g * 8 + 6) * VT_STRIDE + r] = (unsigned short)(d.w & 0xffffu);
      Vt[(g * 8 + 7) * VT_STRIDE + r] = (unsigned short)(d.w >> 16);
    }
    // ---- prefetch next key/value block ----
    if (kb + 1 < nkb) {
      size_t nb = (size_t)(rowbase + (kb + 1) * 64);
      __builtin_prefetch(key +   (nb + (t >> 1)) * D_OUT + h * HEAD_DIM + (t & 1) * 64, 0, 1);
      __builtin_prefetch(value + (nb + (t >> 1)) * D_OUT + h * HEAD_DIM + (t & 1) * 64, 0, 1);
    }
#if USE_ASYNC_LDS
    __builtin_amdgcn_s_wait_asynccnt(0);
#endif
    __syncthreads();

    // ---- S = Q @ K^T : four 16x16 tiles over 64 keys ----
    v8f s[4];
#pragma unroll
    for (int nt = 0; nt < 4; ++nt) {
      s[nt] = v8f{};
#pragma unroll
      for (int c = 0; c < 4; ++c) {
        Frag16 kf;
        const unsigned short* p = &Ks[(nt * 16 + l15) * KS_STRIDE + c * 32];
        kf.q[0] = *(const uint4*)(p + half * 8);
        kf.q[1] = *(const uint4*)(p + 16 + half * 8);
        s[nt] = __builtin_amdgcn_wmma_f32_16x16x32_bf16(
            false, qf[c].v, false, kf.v, (short)0, s[nt], false, false);
      }
    }

    // ---- scale + causal mask + online softmax ----
    const bool diag = (kb == qblk);
    float p[4][8];
#pragma unroll
    for (int nt = 0; nt < 4; ++nt) {
#pragma unroll
      for (int e = 0; e < 8; ++e) {
        float v = s[nt][e] * scale;
        if (diag) {
          int ki = kb * 64 + nt * 16 + l15;
          int qi = qrow0 + e + half * 8;
          if (ki > qi) v = -INFINITY;
        }
        p[nt][e] = v;
      }
    }
    float corr[8];
#pragma unroll
    for (int e = 0; e < 8; ++e) {
      float m = p[0][e];
      m = fmaxf(m, p[1][e]); m = fmaxf(m, p[2][e]); m = fmaxf(m, p[3][e]);
#pragma unroll
      for (int mask = 1; mask < 16; mask <<= 1)
        m = fmaxf(m, __shfl_xor(m, mask, 32));
      float mnew = fmaxf(rmax[e], m);
      corr[e] = __expf(rmax[e] - mnew);
      rmax[e] = mnew;
      float ps = 0.0f;
#pragma unroll
      for (int nt = 0; nt < 4; ++nt) {
        p[nt][e] = __expf(p[nt][e] - mnew);
        ps += p[nt][e];
      }
#pragma unroll
      for (int mask = 1; mask < 16; mask <<= 1)
        ps += __shfl_xor(ps, mask, 32);
      rsum[e] = rsum[e] * corr[e] + ps;
    }
#pragma unroll
    for (int ot = 0; ot < 8; ++ot)
#pragma unroll
      for (int e = 0; e < 8; ++e) o[ot][e] *= corr[e];

    // ---- P (C-fragment order) -> per-wave LDS -> A-fragment order ----
    unsigned short* ps = &Ps[wv][0];
#pragma unroll
    for (int nt = 0; nt < 4; ++nt)
#pragma unroll
      for (int e = 0; e < 8; ++e)
        ps[(e + half * 8) * PS_STRIDE + nt * 16 + l15] = f2bf(p[nt][e]);

    Frag16 pf[2];
#pragma unroll
    for (int kc = 0; kc < 2; ++kc) {
      const unsigned short* pp = &ps[l15 * PS_STRIDE + kc * 32];
      pf[kc].q[0] = *(const uint4*)(pp + half * 8);
      pf[kc].q[1] = *(const uint4*)(pp + 16 + half * 8);
    }

    // ---- O += P @ V ----
#pragma unroll
    for (int kc = 0; kc < 2; ++kc) {
#pragma unroll
      for (int ot = 0; ot < 8; ++ot) {
        Frag16 vf;
        const unsigned short* vp = &Vt[(ot * 16 + l15) * VT_STRIDE + kc * 32];
        vf.q[0] = *(const uint4*)(vp + half * 8);
        vf.q[1] = *(const uint4*)(vp + 16 + half * 8);
        o[ot] = __builtin_amdgcn_wmma_f32_16x16x32_bf16(
            false, pf[kc].v, false, vf.v, (short)0, o[ot], false, false);
      }
    }
  }

  // ---- normalize + store attn output (bf16, (b,s,D_OUT) layout) ----
  unsigned short* op =
      attnout + (size_t)(rowbase + qrow0) * D_OUT + h * HEAD_DIM;
#pragma unroll
  for (int ot = 0; ot < 8; ++ot) {
#pragma unroll
    for (int e = 0; e < 8; ++e) {
      float v = o[ot][e] / rsum[e];
      int m = e + half * 8;
      op[(size_t)m * D_OUT + ot * 16 + l15] = f2bf(v);
    }
  }
}

// ---------------------------------------------------------------------------
// Host launcher
// ---------------------------------------------------------------------------
extern "C" void kernel_launch(void* const* d_in, const int* in_sizes, int n_in,
                              void* d_out, int out_size, void* d_ws, size_t ws_size,
                              hipStream_t stream) {
  const float* x    = (const float*)d_in[0];
  const float* w_kv = (const float*)d_in[1];
  const float* b_kv = (const float*)d_in[2];
  const float* w_ku = (const float*)d_in[3];
  const float* b_ku = (const float*)d_in[4];
  const float* w_kr = (const float*)d_in[5];
  const float* b_kr = (const float*)d_in[6];
  const float* w_vu = (const float*)d_in[7];
  const float* b_vu = (const float*)d_in[8];
  const float* w_q  = (const float*)d_in[9];
  const float* b_q  = (const float*)d_in[10];
  const float* w_qu = (const float*)d_in[11];
  const float* b_qu = (const float*)d_in[12];
  const float* w_qr = (const float*)d_in[13];
  const float* b_qr = (const float*)d_in[14];
  const float* w_o  = (const float*)d_in[15];
  const float* b_o  = (const float*)d_in[16];
  float* out = (float*)d_out;

  // bf16 workspace layout
  unsigned short* ws   = (unsigned short*)d_ws;
  unsigned short* ckv  = ws;                                   // 4096 x 512
  unsigned short* qlat = ckv  + (size_t)NROW * DL;             // 4096 x 512
  unsigned short* keyb = qlat + (size_t)NROW * DL;             // 4096 x 2048
  unsigned short* qry  = keyb + (size_t)NROW * D_OUT;          // 4096 x 2048
  unsigned short* val  = qry  + (size_t)NROW * D_OUT;          // 4096 x 2048
  unsigned short* attn = val  + (size_t)NROW * D_OUT;          // 4096 x 2048

  dim3 blk(256);

  // ckv = x @ w_kv + b_kv          (4096 x 512, K=2048)
  gemm_wmma_bf16<float, unsigned short><<<dim3(NROW/64, DL/256), blk, 0, stream>>>(
      x, D_IN, w_kv, DL, b_kv, ckv, DL, 0, NROW, DL, D_IN);

  // key[:, :1024] = ckv @ w_ku + b_ku   (K=512)
  gemm_wmma_bf16<unsigned short, unsigned short><<<dim3(NROW/64, 1024/256), blk, 0, stream>>>(
      ckv, DL, w_ku, 1024, b_ku, keyb, D_OUT, 0, NROW, 1024, DL);

  // key[:, 1024:] = x @ w_kr + b_kr     (K=2048)
  gemm_wmma_bf16<float, unsigned short><<<dim3(NROW/64, 1024/256), blk, 0, stream>>>(
      x, D_IN, w_kr, 1024, b_kr, keyb, D_OUT, 1024, NROW, 1024, D_IN);

  // q_lat = x @ w_q + b_q               (4096 x 512, K=2048)
  gemm_wmma_bf16<float, unsigned short><<<dim3(NROW/64, DL/256), blk, 0, stream>>>(
      x, D_IN, w_q, DL, b_q, qlat, DL, 0, NROW, DL, D_IN);

  // query[:, :1024] = q_lat @ w_qu + b_qu
  gemm_wmma_bf16<unsigned short, unsigned short><<<dim3(NROW/64, 1024/256), blk, 0, stream>>>(
      qlat, DL, w_qu, 1024, b_qu, qry, D_OUT, 0, NROW, 1024, DL);

  // query[:, 1024:] = q_lat @ w_qr + b_qr
  gemm_wmma_bf16<unsigned short, unsigned short><<<dim3(NROW/64, 1024/256), blk, 0, stream>>>(
      qlat, DL, w_qr, 1024, b_qr, qry, D_OUT, 1024, NROW, 1024, DL);

  // value = ckv @ w_vu + b_vu           (4096 x 2048, K=512)
  gemm_wmma_bf16<unsigned short, unsigned short><<<dim3(NROW/64, D_OUT/256), blk, 0, stream>>>(
      ckv, DL, w_vu, D_OUT, b_vu, val, D_OUT, 0, NROW, D_OUT, DL);

  // fused causal attention -> attn (bf16)
  mla_attention<<<dim3(CTX/64, NUM_HEAD, BATCH), dim3(128), 0, stream>>>(
      qry, keyb, val, attn);

  // out = attn @ w_o + b_o (fp32)       (4096 x 2048, K=2048)
  gemm_wmma_bf16<unsigned short, float><<<dim3(NROW/64, D_OUT/256), blk, 0, stream>>>(
      attn, D_OUT, w_o, D_OUT, b_o, out, D_OUT, 0, NROW, D_OUT, D_OUT);

  (void)n_in; (void)in_sizes; (void)out_size; (void)ws_size;
}
